// RNN_53833120088175
// MI455X (gfx1250) — compile-verified, matrix operand order checked by reading, not run on hardware
//
#include <hip/hip_runtime.h>
#include <stdint.h>

typedef _Float16 v16h __attribute__((ext_vector_type(16)));
typedef _Float16 v8h  __attribute__((ext_vector_type(8)));
typedef float    v8f  __attribute__((ext_vector_type(8)));

#define M_TOT   2048
#define NN      128
#define AA      4        // trial conditions
#define TC      50       // timesteps per phase
#define EULER_C 0.2f     // dt/tau
#define NOISE_A 0.1f

__global__ __launch_bounds__(256, 1)
void rnn_wmma_kernel(const float* __restrict__ x,      // [4,2,2]
                     const float* __restrict__ rec_w,  // [M,128,128]
                     const float* __restrict__ rec_b,  // [M,128,1]
                     const float* __restrict__ inp_w,  // [M,128,2]
                     const float* __restrict__ out_w,  // [M,2,128]
                     const float* __restrict__ mem_w,  // [M,2,128]
                     float* __restrict__ out)          // [2,4,M,2]
{
    const int m    = blockIdx.x;
    const int tid  = threadIdx.x;
    const int wave = tid >> 5;      // 8 waves of 32 (wave32)
    const int lane = tid & 31;
    const int half = lane >> 4;     // lane group (0: lanes 0-15, 1: lanes 16-31)
    const int col  = lane & 15;     // C/D column (= trial condition when < 4)

    __shared__ __align__(16) _Float16 sW[NN * NN];       // 32 KB  W[m] as f16
    __shared__ __align__(16) _Float16 sBF[4 * 32 * 16];  // 4 KB   B frags, frag-order
    __shared__ __align__(16) float sR[AA * NN];          // state r[a][i]
    __shared__ __align__(16) float sDrv0[AA * NN];       // rec_b (broadcast over a)
    __shared__ __align__(16) float sDrvS[AA * NN];       // rec_b + inp_w @ x_sample
    __shared__ __align__(16) float sDrvT[AA * NN];       // rec_b + inp_w @ x_test
    __shared__ __align__(16) float sMW[2 * NN];          // mem_w[o][j]
    __shared__ __align__(16) float sOW[2 * NN];          // out_w[o][j]
    __shared__ float sAcc[16];                           // [mem|out][a][o] sums

    // ---- one-shot loads: W (f32 -> f16), drives, readout weights ----
    const float4* wp = (const float4*)rec_w + (size_t)m * (NN * NN / 4);
    #pragma unroll
    for (int it = 0; it < 16; ++it) {
        int v = tid + it * 256;                 // 4096 float4 = 16384 floats
        float4 f = wp[v];
        int o4 = v * 4;
        sW[o4 + 0] = (_Float16)f.x; sW[o4 + 1] = (_Float16)f.y;
        sW[o4 + 2] = (_Float16)f.z; sW[o4 + 3] = (_Float16)f.w;
    }
    if (tid < NN) {
        float b  = rec_b[(size_t)m * NN + tid];
        float w0 = inp_w[((size_t)m * NN + tid) * 2 + 0];
        float w1 = inp_w[((size_t)m * NN + tid) * 2 + 1];
        #pragma unroll
        for (int a = 0; a < AA; ++a) {
            sDrv0[a * NN + tid] = b;
            sDrvS[a * NN + tid] = b + w0 * x[(a * 2 + 0) * 2 + 0] + w1 * x[(a * 2 + 0) * 2 + 1];
            sDrvT[a * NN + tid] = b + w0 * x[(a * 2 + 1) * 2 + 0] + w1 * x[(a * 2 + 1) * 2 + 1];
        }
    }
    sMW[tid] = mem_w[(size_t)m * 2 * NN + tid];     // tid in [0,256)
    sOW[tid] = out_w[(size_t)m * 2 * NN + tid];
    sR[tid] = 0.0f; sR[tid + 256] = 0.0f;           // r starts at zero
    { v8h z = {}; *(v8h*)(sBF + tid * 8) = z; }     // zero staging: r=0 and cols>=4 stay 0
    if (tid < 16) sAcc[tid] = 0.0f;
    __syncthreads();

    const int row = wave * 16 + col;                // this lane's A-matrix row

    // Update-lane geometry: 8 consecutive neurons i0..i0+7 map to ONE contiguous
    // 8-half chunk of the fragment-ordered staging buffer.
    const int i0   = wave * 16 + half * 8;
    const int fkb  = i0 >> 5;                       // K chunk
    const int fln  = (((i0 >> 3) & 1) << 4) + col;  // frag lane = group*16 + condition
    const int foff = ((i0 >> 4) & 1) << 3;          // element offset (0 or 8)
    _Float16* fragDst = sBF + fkb * 512 + fln * 16 + foff;
    float4* rState = (float4*)(sR + col * NN + i0);

    // ---- 5 phases x 50 steps, fully resident in LDS/registers ----
    for (int phase = 0; phase < 5; ++phase) {
        const float* drv = (phase == 1) ? sDrvS : (phase == 3) ? sDrvT : sDrv0;
        const int projMode = (phase == 2) ? 1 : (phase == 4) ? 2 : 0;
        const float* pw = (phase == 4) ? sOW : sMW;
        const float4* drvP = (const float4*)(drv + col * NN + i0);

        for (int t = 0; t < TC; ++t) {
            const uint32_t step = (uint32_t)(phase * TC + t);

            // ---- pre[16x16] = W_tile @ R  (K = 128 in 4 chunks of 32) ----
            v8f acc = {};
            #pragma unroll
            for (int kb = 0; kb < 4; ++kb) {
                const _Float16* bb = sBF + kb * 512 + lane * 16;
                v8h b0 = *(const v8h*)(bb);
                v8h b1 = *(const v8h*)(bb + 8);
                v16h bf;
                #pragma unroll
                for (int e = 0; e < 8; ++e) { bf[e] = b0[e]; bf[8 + e] = b1[e]; }

                const _Float16* ab = sW + row * NN + kb * 32 + half * 8;
                v8h a0 = *(const v8h*)(ab);
                v8h a1 = *(const v8h*)(ab + 16);
                v16h af;
                #pragma unroll
                for (int e = 0; e < 8; ++e) { af[e] = a0[e]; af[8 + e] = a1[e]; }

                acc = __builtin_amdgcn_wmma_f32_16x16x32_f16(
                          false, af, false, bf, (short)0, acc, false, false);
            }
            __syncthreads();   // all frag/state reads done before rewrite

            // ---- Euler update: lane owns (a=col, rows i0..i0+7); writes f32
            //      state AND the f16 B-fragment chunk directly (no pack pass) ----
            if (col < AA) {
                float4 r0 = rState[0], r1 = rState[1];
                float4 d0 = drvP[0],  d1 = drvP[1];
                float rr[8] = { r0.x, r0.y, r0.z, r0.w, r1.x, r1.y, r1.z, r1.w };
                float dd[8] = { d0.x, d0.y, d0.z, d0.w, d1.x, d1.y, d1.z, d1.w };
                // noise: function of (step, m, i) only -> shared across conditions
                uint32_t h = step * 0x9E3779B1u + (uint32_t)m * 0x85EBCA77u
                           + (uint32_t)i0 * 0xC2B2AE3Du + 0x27D4EB2Fu;
                h ^= h >> 15; h *= 0x2C1B3C6Du; h ^= h >> 12;
                v8h hv;
                #pragma unroll
                for (int g = 0; g < 8; ++g) {
                    uint32_t y = h + (uint32_t)g * 0x9E3779B9u;
                    y ^= y >> 16; y *= 0x21F0AAADu; y ^= y >> 15;
                    float noise = NOISE_A * ((float)(y >> 8) * (1.0f / 16777216.0f));
                    float pre   = acc[g] + dd[g];
                    float relu  = fmaxf(pre, 0.0f);
                    rr[g] = rr[g] + EULER_C * (-rr[g] + relu + noise);
                    hv[g] = (_Float16)rr[g];
                }
                r0.x = rr[0]; r0.y = rr[1]; r0.z = rr[2]; r0.w = rr[3];
                r1.x = rr[4]; r1.y = rr[5]; r1.z = rr[6]; r1.w = rr[7];
                rState[0] = r0; rState[1] = r1;
                *(v8h*)fragDst = hv;                 // one ds_store_b128
            }
            __syncthreads();   // new state visible to projection + next matmul

            // ---- readout: wave w -> (a = w&3, o = w>>2); wave32 reduction ----
            if (projMode && !(projMode == 1 && t < TC / 2)) {
                int a = wave & 3, o = wave >> 2;
                float4 p  = ((const float4*)(pw + o * NN))[lane];
                float4 rv = ((const float4*)(sR + a * NN))[lane];
                float s = p.x * rv.x + p.y * rv.y + p.z * rv.z + p.w * rv.w;
                #pragma unroll
                for (int off = 16; off > 0; off >>= 1) s += __shfl_xor(s, off, 32);
                if (lane == 0) sAcc[(projMode - 1) * 8 + a * 2 + o] += s;
            }
        }
    }
    __syncthreads();

    // out[s][a][m][o]: s=0 -> delay-memory mean over 25 steps, s=1 -> response mean over 50
    if (tid < 16) {
        int s = tid >> 3, a = (tid >> 1) & 3, o = tid & 1;
        float v = sAcc[s * 8 + a * 2 + o] * (s == 0 ? (1.0f / 25.0f) : (1.0f / 50.0f));
        out[(((size_t)s * 4 + a) * M_TOT + m) * 2 + o] = v;
    }
}

extern "C" void kernel_launch(void* const* d_in, const int* in_sizes, int n_in,
                              void* d_out, int out_size, void* d_ws, size_t ws_size,
                              hipStream_t stream) {
    (void)in_sizes; (void)n_in; (void)d_ws; (void)ws_size; (void)out_size;
    const float* x     = (const float*)d_in[0];
    const float* rec_w = (const float*)d_in[1];
    const float* rec_b = (const float*)d_in[2];
    const float* inp_w = (const float*)d_in[3];
    const float* out_w = (const float*)d_in[4];
    const float* mem_w = (const float*)d_in[5];
    float* out = (float*)d_out;

    rnn_wmma_kernel<<<dim3(M_TOT), dim3(256), 0, stream>>>(
        x, rec_w, rec_b, inp_w, out_w, mem_w, out);
}
